// TopicLanguageModel_53463752900645
// MI455X (gfx1250) — matrix operand order repaired and checked by reading, not compiled, not run on hardware
//
#include <hip/hip_runtime.h>
#include <hip/hip_bf16.h>

// ---------------------------------------------------------------------------
// TopicLanguageModel on MI455X (gfx1250, wave32, WMMA bf16)
//
//   logits[B,S,V] = LSTM(concat(topic, embed[tokens])) @ dense_w + dense_b
//
// Pipeline:
//   1) pack_b_frags : Wx / Wh / dense_w  f32 -> bf16 WMMA B-fragments
//   2) pack_x_frags : gather+concat xs   f32 -> bf16 WMMA A-fragments
//   3) gemm_frag    : preact[4096,2048] = xs @ Wx + bias   (parallel, 9.7 GF)
//   4) lstm_seq     : 256 sequential steps, 1 WGP, h in LDS A-frag layout,
//                     c in registers, recurrent GEMM via v_wmma bf16
//   5) gemm_frag    : logits[4096,32000] = hs @ Wd + b     (parallel, 134 GF)
// ---------------------------------------------------------------------------

typedef __attribute__((ext_vector_type(16))) __bf16 v16bf;
typedef __attribute__((ext_vector_type(8)))  float  v8f;

#define B_  16
#define S_  256
#define V_  32000
#define D_  512
#define H_  512
#define T_  50

#define KX_TILES 18    // K = T+D = 562 padded to 576 = 18*32
#define KH_TILES 16    // K = H = 512 = 16*32
#define NT_Z     128   // N = 4H = 2048 -> 128 n-tiles
#define NT_V     2000  // N = V = 32000 -> 2000 n-tiles
#define MT_X     256   // M = B*S = 4096 -> 256 m-tiles

// 16-bit A-matrix (16x32) lane/element -> K mapping (ISA 7.12.2):
// lane<16 : e0..7 -> K0..7,  e8..15 -> K16..23
// lane>=16: e0..7 -> K8..15, e8..15 -> K24..31
static __device__ __forceinline__ int kmapA(int e, int half) {
    return (e < 8 ? e : e + 8) + 8 * half;
}

// ---------------------------------------------------------------------------
// Pack a [K x N] row-major f32 matrix into bf16 WMMA B-fragments
// dst layout: [NT][KT][32 lanes][16 elems]
// B-matrix (32x16) mapping: same K order as A (kmapA): K=kt*32+kmapA(e, lane>>4)
// ---------------------------------------------------------------------------
__global__ __launch_bounds__(256) void pack_b_frags(
    const float* __restrict__ src, int src_ld, int row_off, int krows,
    __bf16* __restrict__ dst, int NT, int KT, int ncols)
{
    size_t total = (size_t)NT * KT * 512;
    size_t i = (size_t)blockIdx.x * blockDim.x + threadIdx.x;
    if (i >= total) return;
    int e    = (int)(i & 15);
    int lane = (int)((i >> 4) & 31);
    int kt   = (int)((i >> 9) % (size_t)KT);
    int nt   = (int)(i / ((size_t)KT * 512));
    int half = lane >> 4, l4 = lane & 15;
    int k = kt * 32 + kmapA(e, half);
    int n = nt * 16 + l4;
    float v = 0.0f;
    if (k < krows && n < ncols) v = src[(size_t)(row_off + k) * src_ld + n];
    dst[i] = (__bf16)v;
}

// ---------------------------------------------------------------------------
// Build xs = concat(topic_probs, embed[tokens]) as bf16 WMMA A-fragments
// dst layout: [MT_X][KX_TILES][32 lanes][16 elems], rows m = b*S + s
// ---------------------------------------------------------------------------
__global__ __launch_bounds__(256) void pack_x_frags(
    const int* __restrict__ tokens, const float* __restrict__ topic,
    const float* __restrict__ embed, __bf16* __restrict__ dst)
{
    size_t total = (size_t)MT_X * KX_TILES * 512;
    size_t i = (size_t)blockIdx.x * blockDim.x + threadIdx.x;
    if (i >= total) return;
    int e    = (int)(i & 15);
    int lane = (int)((i >> 4) & 31);
    int kt   = (int)((i >> 9) % (size_t)KX_TILES);
    int mt   = (int)(i / ((size_t)KX_TILES * 512));
    int half = lane >> 4, l4 = lane & 15;
    int k = kt * 32 + kmapA(e, half);
    int m = mt * 16 + l4;            // global row = b*S + s
    int b = m >> 8, s = m & (S_ - 1);
    float v = 0.0f;
    if (k < T_) {
        v = topic[((size_t)(b * S_ + s)) * T_ + k];
    } else if (k < T_ + D_) {
        int tok = tokens[b * S_ + s];
        v = embed[(size_t)tok * D_ + (k - T_)];
    }
    dst[i] = (__bf16)v;
}

// ---------------------------------------------------------------------------
// Fragment GEMM: out[M, nld] (f32) = A_frag @ B_frag + bias
// Each wave computes a 4x4 block of 16x16 tiles (16 v_wmma per K-step).
// ---------------------------------------------------------------------------
__global__ __launch_bounds__(256) void gemm_frag(
    const __bf16* __restrict__ A, const __bf16* __restrict__ Bm,
    const float* __restrict__ bias, float* __restrict__ out,
    int MT, int NT, int KT, int nld)
{
    int wid  = threadIdx.x >> 5;
    int lane = threadIdx.x & 31;
    int gw   = blockIdx.x * 8 + wid;
    int nblks = NT >> 2, mblks = MT >> 2;
    if (gw >= mblks * nblks) return;
    int mb = gw / nblks, nb = gw % nblks;
    int mt0 = mb * 4, nt0 = nb * 4;

    v8f zero = {0.f, 0.f, 0.f, 0.f, 0.f, 0.f, 0.f, 0.f};
    v8f acc[4][4];
#pragma unroll
    for (int i = 0; i < 4; ++i)
#pragma unroll
        for (int j = 0; j < 4; ++j) acc[i][j] = zero;

    for (int kt = 0; kt < KT; ++kt) {
        v16bf af[4], bf[4];
#pragma unroll
        for (int i = 0; i < 4; ++i)
            af[i] = *(const v16bf*)(A + (((size_t)(mt0 + i) * KT + kt) * 32 + lane) * 16);
#pragma unroll
        for (int j = 0; j < 4; ++j)
            bf[j] = *(const v16bf*)(Bm + (((size_t)(nt0 + j) * KT + kt) * 32 + lane) * 16);
#pragma unroll
        for (int i = 0; i < 4; ++i)
#pragma unroll
            for (int j = 0; j < 4; ++j)
                acc[i][j] = __builtin_amdgcn_wmma_f32_16x16x32_bf16(
                    false, af[i], false, bf[j], (short)0, acc[i][j], false, false);
    }

    int half = lane >> 4, l4 = lane & 15;
#pragma unroll
    for (int i = 0; i < 4; ++i) {
#pragma unroll
        for (int j = 0; j < 4; ++j) {
            int n  = (nt0 + j) * 16 + l4;
            float bv = bias ? bias[n] : 0.0f;
#pragma unroll
            for (int v = 0; v < 8; ++v) {
                int m = (mt0 + i) * 16 + v + 8 * half;
                out[(size_t)m * nld + n] = acc[i][j][v] + bv;
            }
        }
    }
}

// ---------------------------------------------------------------------------
// Sequential LSTM: one workgroup (32 wave32 waves on one WGP).
// Wave w owns h-columns [w*16, w*16+16): its 4 z-tiles (i/j/f/o) and the
// cell state c (registers). h lives in LDS already in A-fragment layout so
// every wave's WMMA A-operand is a contiguous 32B ds_load.
// ---------------------------------------------------------------------------
__global__ __launch_bounds__(1024) void lstm_seq(
    const float* __restrict__ pre,     // [4096, 2048] rows = b*S + s
    const __bf16* __restrict__ WhF,    // [NT_Z][16][32][16]
    __bf16* __restrict__ hsF)          // [MT_X][16][32][16]  (A-frags of hs)
{
    __shared__ __align__(128) __bf16 hfrag[16][32][16];   // 16 KB

    int tid  = threadIdx.x;
    int wid  = tid >> 5;
    int lane = tid & 31;
    int half = lane >> 4, l4 = lane & 15;

    for (int i = tid; i < 16 * 32 * 16; i += 1024)
        ((__bf16*)hfrag)[i] = (__bf16)0.0f;

    float cst[8];
#pragma unroll
    for (int v = 0; v < 8; ++v) cst[v] = 0.0f;
    __syncthreads();

    for (int s = 0; s < S_; ++s) {
        v8f acc[4];
        // z-tile init from precomputed input projection (+bias already folded)
#pragma unroll
        for (int g = 0; g < 4; ++g) {
            int n = (g * 32 + wid) * 16 + l4;
#pragma unroll
            for (int v = 0; v < 8; ++v) {
                int m = v + 8 * half;                       // batch row
                acc[g][v] = pre[((size_t)(m * S_ + s)) * 2048 + n];
            }
        }
        // recurrent h @ Wh  (K = 512 -> 16 WMMA k-steps, 4 gate tiles)
        for (int kt = 0; kt < 16; ++kt) {
            v16bf a = *(const v16bf*)&hfrag[kt][lane][0];
#pragma unroll
            for (int g = 0; g < 4; ++g) {
                v16bf b = *(const v16bf*)(WhF + (((size_t)(g * 32 + wid) * 16 + kt) * 32 + lane) * 16);
                acc[g] = __builtin_amdgcn_wmma_f32_16x16x32_bf16(
                    false, a, false, b, (short)0, acc[g], false, false);
            }
        }
        __syncthreads();   // everyone done reading hfrag before we rewrite it

        // gates (TF order i, j, f, o; forget_bias = 0)
#pragma unroll
        for (int v = 0; v < 8; ++v) {
            float iv = 1.0f / (1.0f + __expf(-acc[0][v]));
            float jv = tanhf(acc[1][v]);
            float fv = 1.0f / (1.0f + __expf(-acc[2][v]));
            float ov = 1.0f / (1.0f + __expf(-acc[3][v]));
            cst[v] = fv * cst[v] + iv * jv;
            float h = ov * tanhf(cst[v]);

            int m  = v + 8 * half;            // batch row 0..15
            int hc = wid * 16 + l4;           // h column 0..511
            int kt2 = hc >> 5, ko = hc & 31;
            int half2 = (ko >> 3) & 1;
            int e2 = (ko & 16) ? (ko & 7) + 8 : (ko & 7);  // inverse kmapA
            __bf16 hb = (__bf16)h;
            hfrag[kt2][half2 * 16 + m][e2] = hb;           // next-step A frag

            int row = m * S_ + s;                          // hs A-frag row
            int mt = row >> 4, l4r = row & 15;
            hsF[(((size_t)mt * 16 + kt2) * 32 + half2 * 16 + l4r) * 16 + e2] = hb;
        }
        __syncthreads();   // hfrag ready for next step
    }
}

// ---------------------------------------------------------------------------
// Host launcher
// ---------------------------------------------------------------------------
extern "C" void kernel_launch(void* const* d_in, const int* in_sizes, int n_in,
                              void* d_out, int out_size, void* d_ws, size_t ws_size,
                              hipStream_t stream) {
    const int*   tokens = (const int*)d_in[0];     // [B,S]
    const float* topic  = (const float*)d_in[1];   // [B,S,T]
    const float* embed  = (const float*)d_in[2];   // [V,D]
    const float* lker   = (const float*)d_in[3];   // [T+D+H, 4H]
    const float* lbias  = (const float*)d_in[4];   // [4H]
    const float* dw     = (const float*)d_in[5];   // [H,V]
    const float* db     = (const float*)d_in[6];   // [V]
    float* out = (float*)d_out;                    // [B,S,V] f32

    char* ws = (char*)d_ws;
    size_t off = 0;
    auto take = [&](size_t bytes) -> char* {
        char* p = ws + off;
        off += (bytes + 255) & ~(size_t)255;
        return p;
    };
    __bf16* WxF = (__bf16*)take((size_t)NT_Z * KX_TILES * 512 * 2);  // 2.4 MB
    __bf16* WhF = (__bf16*)take((size_t)NT_Z * KH_TILES * 512 * 2);  // 2.1 MB
    __bf16* WdF = (__bf16*)take((size_t)NT_V * KH_TILES * 512 * 2);  // 32.8 MB
    __bf16* XF  = (__bf16*)take((size_t)MT_X * KX_TILES * 512 * 2);  // 4.7 MB
    float*  PRE = (float*) take((size_t)(B_ * S_) * 2048 * 4);       // 33.6 MB
    __bf16* HSF = (__bf16*)take((size_t)MT_X * KH_TILES * 512 * 2);  // 4.2 MB

    // 1) weight packing (f32 -> bf16 WMMA B-fragments)
    {
        size_t tot = (size_t)NT_Z * KX_TILES * 512;
        pack_b_frags<<<(unsigned)((tot + 255) / 256), 256, 0, stream>>>(
            lker, 2048, 0, T_ + D_, WxF, NT_Z, KX_TILES, 2048);
    }
    {
        size_t tot = (size_t)NT_Z * KH_TILES * 512;
        pack_b_frags<<<(unsigned)((tot + 255) / 256), 256, 0, stream>>>(
            lker, 2048, T_ + D_, H_, WhF, NT_Z, KH_TILES, 2048);
    }
    {
        size_t tot = (size_t)NT_V * KH_TILES * 512;
        pack_b_frags<<<(unsigned)((tot + 255) / 256), 256, 0, stream>>>(
            dw, V_, 0, H_, WdF, NT_V, KH_TILES, V_);
    }
    // 2) gather + concat xs into A-fragments
    {
        size_t tot = (size_t)MT_X * KX_TILES * 512;
        pack_x_frags<<<(unsigned)((tot + 255) / 256), 256, 0, stream>>>(
            tokens, topic, embed, XF);
    }
    // 3) input projection: PRE = xs @ Wx + bias  (parallel GEMM)
    {
        int waves = (MT_X / 4) * (NT_Z / 4);
        gemm_frag<<<(unsigned)((waves + 7) / 8), 256, 0, stream>>>(
            XF, WxF, lbias, PRE, MT_X, NT_Z, KX_TILES, 2048);
    }
    // 4) sequential LSTM recurrence (single WGP)
    lstm_seq<<<1, 1024, 0, stream>>>(PRE, WhF, HSF);
    // 5) logits = hs @ dense_w + dense_b  (dominant 134 GFLOP GEMM)
    {
        int waves = (MT_X / 4) * (NT_V / 4);
        gemm_frag<<<(unsigned)((waves + 7) / 8), 256, 0, stream>>>(
            HSF, WdF, db, out, MT_X, NT_V, KH_TILES, V_);
    }
}
